// Self_Attn_3435973837459
// MI455X (gfx1250) — matrix operand order repaired and checked by reading, not compiled, not run on hardware
//
#include <hip/hip_runtime.h>

// ---------------------------------------------------------------------------
// Self_Attn channel-attention block for MI455X (gfx1250, wave32, WMMA).
// All GEMMs use v_wmma_f32_16x16x32_bf16 (bf16 A/B, f32 accumulate).
// Staging is software-pipelined: next K-step tiles are fetched into registers
// while the current K-step's WMMA runs (hides global latency behind matrix ops).
// ---------------------------------------------------------------------------

#define CCH   256
#define NPIX  4096          // H*W
#define BATCH 16
#define PTOT  65536         // BATCH*NPIX
#define NHEAD 8

typedef __attribute__((ext_vector_type(16))) __bf16       v16bf;
typedef __attribute__((ext_vector_type(8)))  float        v8f;
typedef __attribute__((ext_vector_type(4)))  unsigned int v4u;
typedef __attribute__((ext_vector_type(2)))  unsigned int v2u;

union Frag { v16bf v; v4u q[2]; };
union U16x4 { v2u d; unsigned short s[4]; };
union U16x8 { v4u d; unsigned short s[8]; };

__device__ __forceinline__ unsigned short f2bf(float f) {
  unsigned int u = __float_as_uint(f);
  u += 0x7FFFu + ((u >> 16) & 1u);   // round-to-nearest-even
  return (unsigned short)(u >> 16);
}

__device__ __forceinline__ v8f wmma_bf16(const Frag& a, const Frag& b, v8f c) {
  // D = A(16x32 bf16) * B(32x16 bf16) + C(16x16 f32)
  return __builtin_amdgcn_wmma_f32_16x16x32_bf16(
      /*neg_a=*/false, a.v, /*neg_b=*/false, b.v,
      /*c_mod=*/(short)0, c, /*reuse_a=*/false, /*reuse_b=*/false);
}

// ---------------------------------------------------------------------------
// f32 -> bf16 conversion
// ---------------------------------------------------------------------------
__global__ void cvt_f32_bf16(const float* __restrict__ src,
                             unsigned short* __restrict__ dst, int n) {
  int i = blockIdx.x * 256 + threadIdx.x;
  if (i < n) dst[i] = f2bf(src[i]);
}

// ---------------------------------------------------------------------------
// One power iteration: rsig = 1 / || W * l2(W^T u) ||   (W is 256x256 f32)
// ---------------------------------------------------------------------------
__global__ void spectral_sigma(const float* __restrict__ W,
                               const float* __restrict__ u,
                               float* __restrict__ rsig_out) {
  __shared__ float vv[CCH];
  __shared__ float red[CCH];
  int t = threadIdx.x;

  float a = 0.f;                                   // v_j = sum_o W[o][j]*u[o]
  for (int o = 0; o < CCH; ++o) a += W[o * CCH + t] * u[o];
  vv[t] = a;
  red[t] = a * a;
  __syncthreads();
  for (int s = 128; s > 0; s >>= 1) {
    if (t < s) red[t] += red[t + s];
    __syncthreads();
  }
  float inv = rsqrtf(red[0] + 1e-24f);
  __syncthreads();
  float vj = vv[t] * inv;                          // l2 normalize
  __syncthreads();
  vv[t] = vj;
  __syncthreads();

  float b = 0.f;                                   // (W v)_o
  for (int j = 0; j < CCH; ++j) b += W[t * CCH + j] * vv[j];
  red[t] = b * b;
  __syncthreads();
  for (int s = 128; s > 0; s >>= 1) {
    if (t < s) red[t] += red[t + s];
    __syncthreads();
  }
  if (t == 0) rsig_out[0] = rsqrtf(red[0] + 1e-24f);  // 1/sigma
}

// ---------------------------------------------------------------------------
// Channel GEMM: Out[o, p] = sum_c W[o,c] * X[c, p], p = b*4096 + s.
// X/Out layout: (B, C, N) i.e. addr = b*1048576 + c*4096 + s  (bf16 u16).
// 256 threads = 8 waves, each wave does one 16x16 tile -> block tile 128x16.
//   MODE 0: 3 weights fused (Q,K,V), epilogue + bias -> bf16 O0..O2
//   MODE 1: 1 weight, epilogue acc*rsig + b, leaky_relu(0.1) -> bf16 O0
//   MODE 2: 1 weight, epilogue acc*rsig + b, outf = out1f + beta*y  (f32)
// ---------------------------------------------------------------------------
template <int MODE>
__global__ __launch_bounds__(256)
void gemm_cproj(const unsigned short* __restrict__ X,
                const unsigned short* __restrict__ Wa,
                const unsigned short* __restrict__ Wb,
                const unsigned short* __restrict__ Wc,
                const float* __restrict__ b0, const float* __restrict__ b1,
                const float* __restrict__ b2,
                unsigned short* __restrict__ O0, unsigned short* __restrict__ O1,
                unsigned short* __restrict__ O2,
                const float* __restrict__ rsig,
                const float* __restrict__ beta,
                const float* __restrict__ out1f, float* __restrict__ outf) {
  constexpr int NW = (MODE == 0) ? 3 : 1;
  __shared__ __align__(16) unsigned short ldsA[8][NW][16][32];
  __shared__ __align__(16) unsigned short ldsBt[16][32];   // [n][k]

  const int t   = threadIdx.x;
  const int wid = t >> 5;
  const int l   = t & 31;
  const int rr  = l & 15;
  const int hh  = l >> 4;

  const int bx = blockIdx.x;
  const int bidx = bx >> 8;                 // batch
  const int s0   = (bx & 255) << 4;         // 16-wide pixel tile
  const size_t colbase = (size_t)bidx * ((size_t)CCH * NPIX) + (size_t)s0;
  const int obase = blockIdx.y * 128 + wid * 16;

  // staging indices
  const int kk = t >> 2;                    // B row (t<128 only)
  const int n0 = (t & 3) * 4;
  const unsigned short* wr0base = Wa + (size_t)(obase + rr) * CCH + hh * 16;
  const unsigned short* wr1base = Wb + (size_t)(obase + rr) * CCH + hh * 16;
  const unsigned short* wr2base = Wc + (size_t)(obase + rr) * CCH + hh * 16;

  v2u bstage = {};
  v4u a0s0 = {}, a0s1 = {}, a1s0 = {}, a1s1 = {}, a2s0 = {}, a2s1 = {};

  auto stage_load = [&](int k0) {
    if (t < 128) {
      bstage = *(const v2u*)(X + colbase + (size_t)(k0 + kk) * NPIX + n0);
      if (k0 + 32 < CCH)   // warm next tile in WGP$/L2 (global_prefetch_b8)
        __builtin_prefetch(X + colbase + (size_t)(k0 + 32 + kk) * NPIX + n0, 0, 0);
    }
    a0s0 = *(const v4u*)(wr0base + k0);
    a0s1 = *(const v4u*)(wr0base + k0 + 8);
    if constexpr (NW == 3) {
      a1s0 = *(const v4u*)(wr1base + k0);
      a1s1 = *(const v4u*)(wr1base + k0 + 8);
      a2s0 = *(const v4u*)(wr2base + k0);
      a2s1 = *(const v4u*)(wr2base + k0 + 8);
    }
  };

  const v8f vzero = {0.f, 0.f, 0.f, 0.f, 0.f, 0.f, 0.f, 0.f};
  v8f acc0 = vzero, acc1 = vzero, acc2 = vzero;

  stage_load(0);
  for (int k0 = 0; k0 < CCH; k0 += 32) {
    __syncthreads();                    // previous K-step's LDS reads done
    // ---- commit staged tiles to LDS ----
    if (t < 128) {
      U16x4 tmp;
      tmp.d = bstage;
#pragma unroll
      for (int i = 0; i < 4; ++i) ldsBt[n0 + i][kk] = tmp.s[i];
    }
    *(v4u*)&ldsA[wid][0][rr][hh * 16]     = a0s0;
    *(v4u*)&ldsA[wid][0][rr][hh * 16 + 8] = a0s1;
    if constexpr (NW == 3) {
      *(v4u*)&ldsA[wid][1][rr][hh * 16]     = a1s0;
      *(v4u*)&ldsA[wid][1][rr][hh * 16 + 8] = a1s1;
      *(v4u*)&ldsA[wid][2][rr][hh * 16]     = a2s0;
      *(v4u*)&ldsA[wid][2][rr][hh * 16 + 8] = a2s1;
    }
    __syncthreads();                    // staged tiles visible
    // ---- prefetch next K-step into registers (overlaps with WMMA below) ----
    if (k0 + 32 < CCH) stage_load(k0 + 32);
    // ---- fragments + WMMA ----
    Frag bf;
    bf.q[0] = *(const v4u*)&ldsBt[rr][hh * 16];
    bf.q[1] = *(const v4u*)&ldsBt[rr][hh * 16 + 8];
    {
      Frag af;
      af.q[0] = *(const v4u*)&ldsA[wid][0][rr][hh * 16];
      af.q[1] = *(const v4u*)&ldsA[wid][0][rr][hh * 16 + 8];
      acc0 = wmma_bf16(af, bf, acc0);
    }
    if constexpr (NW == 3) {
      Frag af;
      af.q[0] = *(const v4u*)&ldsA[wid][1][rr][hh * 16];
      af.q[1] = *(const v4u*)&ldsA[wid][1][rr][hh * 16 + 8];
      acc1 = wmma_bf16(af, bf, acc1);
      Frag ag;
      ag.q[0] = *(const v4u*)&ldsA[wid][2][rr][hh * 16];
      ag.q[1] = *(const v4u*)&ldsA[wid][2][rr][hh * 16 + 8];
      acc2 = wmma_bf16(ag, bf, acc2);
    }
  }

  // ---- epilogue (C layout: lane n = l&15; row m = r + 8*(l>>4)) ----
  const int n = rr;
  if constexpr (MODE == 0) {
#pragma unroll
    for (int r = 0; r < 8; ++r) {
      int o = obase + r + 8 * hh;
      size_t idx = colbase + (size_t)o * NPIX + n;
      O0[idx] = f2bf(acc0[r] + b0[o]);
      O1[idx] = f2bf(acc1[r] + b1[o]);
      O2[idx] = f2bf(acc2[r] + b2[o]);
    }
  } else if constexpr (MODE == 1) {
    float rs = rsig[0];
#pragma unroll
    for (int r = 0; r < 8; ++r) {
      int o = obase + r + 8 * hh;
      size_t idx = colbase + (size_t)o * NPIX + n;
      float y = acc0[r] * rs + b0[o];
      y = (y > 0.f) ? y : 0.1f * y;                 // leaky_relu(0.1)
      O0[idx] = f2bf(y);
    }
  } else {
    float rs = rsig[0];
    float bet = beta[0];
#pragma unroll
    for (int r = 0; r < 8; ++r) {
      int o = obase + r + 8 * hh;
      size_t idx = colbase + (size_t)o * NPIX + n;
      float y = acc0[r] * rs + b0[o];
      outf[idx] = out1f[idx] + bet * y;             // final residual
    }
  }
}

// ---------------------------------------------------------------------------
// energy[b,h,i,j] = scale * sum_s q[..i,s]*k[..j,s];  softmax over j.
// One block per (b,h): 128 threads = 4 waves, each a 16x16 tile of 32x32.
// Software-pipelined staging over the 128 K-steps.
// ---------------------------------------------------------------------------
__global__ __launch_bounds__(128)
void attn_energy(const unsigned short* __restrict__ qb,
                 const unsigned short* __restrict__ kb,
                 float* __restrict__ attn_out,
                 unsigned short* __restrict__ attnb) {
  __shared__ __align__(16) unsigned short ldsQ[32][32];  // [i][s-chunk]
  __shared__ __align__(16) unsigned short ldsK[32][32];  // [j][s-chunk]
  __shared__ float eng[32][33];

  const int bh = blockIdx.x;
  const int b  = bh >> 3;
  const int h  = bh & 7;
  const size_t base = (size_t)b * ((size_t)CCH * NPIX) + (size_t)h * 32 * NPIX;

  const int t   = threadIdx.x;
  const int wid = t >> 5;
  const int l   = t & 31;
  const int rr  = l & 15;
  const int hh  = l >> 4;
  const int mi  = wid >> 1;
  const int ni  = wid & 1;

  const int r  = t >> 2;            // staging row
  const int q8 = (t & 3) * 8;
  const unsigned short* qrow = qb + base + (size_t)r * NPIX + q8;
  const unsigned short* krow = kb + base + (size_t)r * NPIX + q8;

  v4u qstage = {}, kstage = {};
  auto stage_load = [&](int k0) {
    qstage = *(const v4u*)(qrow + k0);
    kstage = *(const v4u*)(krow + k0);
    if (k0 + 32 < NPIX) {           // global_prefetch_b8 two steps ahead
      __builtin_prefetch(qrow + k0 + 32, 0, 0);
      __builtin_prefetch(krow + k0 + 32, 0, 0);
    }
  };

  const v8f vzero = {0.f, 0.f, 0.f, 0.f, 0.f, 0.f, 0.f, 0.f};
  v8f acc = vzero;

  stage_load(0);
  for (int k0 = 0; k0 < NPIX; k0 += 32) {
    __syncthreads();
    *(v4u*)&ldsQ[r][q8] = qstage;
    *(v4u*)&ldsK[r][q8] = kstage;
    __syncthreads();
    if (k0 + 32 < NPIX) stage_load(k0 + 32);
    Frag af, bf;
    af.q[0] = *(const v4u*)&ldsQ[mi * 16 + rr][hh * 16];
    af.q[1] = *(const v4u*)&ldsQ[mi * 16 + rr][hh * 16 + 8];
    bf.q[0] = *(const v4u*)&ldsK[ni * 16 + rr][hh * 16];
    bf.q[1] = *(const v4u*)&ldsK[ni * 16 + rr][hh * 16 + 8];
    acc = wmma_bf16(af, bf, acc);
  }

  const float scale = 0.35355339059327373f;  // 8^-0.5
#pragma unroll
  for (int r2 = 0; r2 < 8; ++r2) {
    eng[mi * 16 + r2 + 8 * hh][ni * 16 + rr] = acc[r2] * scale;
  }
  __syncthreads();

  if (t < 32) {                               // softmax over j, one row each
    float row[32];
    float m = -3.4e38f;
#pragma unroll
    for (int j = 0; j < 32; ++j) { row[j] = eng[t][j]; m = fmaxf(m, row[j]); }
    float s = 0.f;
#pragma unroll
    for (int j = 0; j < 32; ++j) { row[j] = __expf(row[j] - m); s += row[j]; }
    float inv = 1.f / s;
    size_t ob = (size_t)bh * 1024 + (size_t)t * 32;
#pragma unroll
    for (int j = 0; j < 32; ++j) {
      float a = row[j] * inv;
      attn_out[ob + j] = a;
      attnb[ob + j] = f2bf(a);
    }
  }
}

// ---------------------------------------------------------------------------
// out1 = gamma * (attn @ v) + x.   Per (b,h), s-blocks of 32.
// 128 threads = 4 waves; single K=32 WMMA step per tile.
// ---------------------------------------------------------------------------
__global__ __launch_bounds__(128)
void attn_apply(const unsigned short* __restrict__ attnb,
                const unsigned short* __restrict__ vb,
                const float* __restrict__ x,
                const float* __restrict__ gamma,
                float* __restrict__ out1,
                unsigned short* __restrict__ out1b) {
  __shared__ __align__(16) unsigned short A[32][32];    // attn [i][j]
  __shared__ __align__(16) unsigned short Bt[32][32];   // [s][j]

  const int bh = blockIdx.y;
  const int b  = bh >> 3;
  const int h  = bh & 7;
  const int s0 = blockIdx.x * 32;
  const size_t vbase = (size_t)b * ((size_t)CCH * NPIX) + (size_t)h * 32 * NPIX;

  const int t  = threadIdx.x;
  const int wid = t >> 5;
  const int l   = t & 31;
  const int rr  = l & 15;
  const int hh  = l >> 4;

  {
    int r  = t >> 2;
    int q8 = (t & 3) * 8;
    *(v4u*)&A[r][q8] = *(const v4u*)(attnb + (size_t)bh * 1024 + r * 32 + q8);
  }
  {
    int j  = t >> 2;
    int n0 = (t & 3) * 8;
    U16x8 tmp;
    tmp.d = *(const v4u*)(vb + vbase + (size_t)j * NPIX + s0 + n0);
#pragma unroll
    for (int i = 0; i < 8; ++i) Bt[n0 + i][j] = tmp.s[i];
  }
  __syncthreads();

  const int mi = wid >> 1;
  const int ni = wid & 1;
  Frag af, bf;
  af.q[0] = *(const v4u*)&A[mi * 16 + rr][hh * 16];
  af.q[1] = *(const v4u*)&A[mi * 16 + rr][hh * 16 + 8];
  bf.q[0] = *(const v4u*)&Bt[ni * 16 + rr][hh * 16];
  bf.q[1] = *(const v4u*)&Bt[ni * 16 + rr][hh * 16 + 8];

  const v8f vzero = {0.f, 0.f, 0.f, 0.f, 0.f, 0.f, 0.f, 0.f};
  v8f acc = wmma_bf16(af, bf, vzero);

  float g = gamma[0];
#pragma unroll
  for (int r = 0; r < 8; ++r) {
    int i = mi * 16 + r + 8 * hh;
    int s = s0 + ni * 16 + rr;
    size_t idx = vbase + (size_t)i * NPIX + s;
    float o = g * acc[r] + x[idx];
    out1[idx] = o;
    out1b[idx] = f2bf(o);
  }
}

// ---------------------------------------------------------------------------
// Workspace layout (bytes)
// ---------------------------------------------------------------------------
static constexpr size_t SZ_WB   = (size_t)CCH * CCH * 2;        // 131072
static constexpr size_t SZ_PC2  = (size_t)PTOT * CCH * 2;       // 33554432
static constexpr size_t OFF_RSIG  = 0;
static constexpr size_t OFF_WQB   = 256;
static constexpr size_t OFF_WKB   = OFF_WQB + SZ_WB;
static constexpr size_t OFF_WVB   = OFF_WKB + SZ_WB;
static constexpr size_t OFF_W1B   = OFF_WVB + SZ_WB;
static constexpr size_t OFF_W2B   = OFF_W1B + SZ_WB;
static constexpr size_t OFF_XB    = OFF_W2B + SZ_WB;
static constexpr size_t OFF_QB    = OFF_XB + SZ_PC2;
static constexpr size_t OFF_KB    = OFF_QB + SZ_PC2;
static constexpr size_t OFF_VB    = OFF_KB + SZ_PC2;
static constexpr size_t OFF_OUT1B = OFF_VB + SZ_PC2;
static constexpr size_t OFF_Y1B   = OFF_OUT1B + SZ_PC2;
static constexpr size_t OFF_ATTNB = OFF_Y1B + SZ_PC2;
static constexpr size_t OFF_OUT1F = OFF_ATTNB + (size_t)BATCH * NHEAD * 32 * 32 * 2;

extern "C" void kernel_launch(void* const* d_in, const int* in_sizes, int n_in,
                              void* d_out, int out_size, void* d_ws, size_t ws_size,
                              hipStream_t stream) {
  (void)in_sizes; (void)n_in; (void)out_size; (void)ws_size;

  const float* x   = (const float*)d_in[0];
  const float* Wq  = (const float*)d_in[1];
  const float* bq  = (const float*)d_in[2];
  const float* Wk  = (const float*)d_in[3];
  const float* bk  = (const float*)d_in[4];
  const float* Wv  = (const float*)d_in[5];
  const float* bv  = (const float*)d_in[6];
  const float* gamma = (const float*)d_in[7];
  const float* beta  = (const float*)d_in[8];
  const float* W1  = (const float*)d_in[9];
  const float* b1  = (const float*)d_in[10];
  const float* u1  = (const float*)d_in[11];
  const float* W2  = (const float*)d_in[12];
  const float* b2  = (const float*)d_in[13];
  const float* u2  = (const float*)d_in[14];

  char* ws = (char*)d_ws;
  float* rsig = (float*)(ws + OFF_RSIG);
  unsigned short* wqb   = (unsigned short*)(ws + OFF_WQB);
  unsigned short* wkb   = (unsigned short*)(ws + OFF_WKB);
  unsigned short* wvb   = (unsigned short*)(ws + OFF_WVB);
  unsigned short* w1b   = (unsigned short*)(ws + OFF_W1B);
  unsigned short* w2b   = (unsigned short*)(ws + OFF_W2B);
  unsigned short* xb    = (unsigned short*)(ws + OFF_XB);
  unsigned short* qb    = (unsigned short*)(ws + OFF_QB);
  unsigned short* kb    = (unsigned short*)(ws + OFF_KB);
  unsigned short* vbuf  = (unsigned short*)(ws + OFF_VB);
  unsigned short* out1b = (unsigned short*)(ws + OFF_OUT1B);
  unsigned short* y1b   = (unsigned short*)(ws + OFF_Y1B);
  unsigned short* attnb = (unsigned short*)(ws + OFF_ATTNB);
  float* out1f = (float*)(ws + OFF_OUT1F);

  float* outf = (float*)d_out;
  float* attn_out = outf + (size_t)BATCH * CCH * NPIX;

  // 1) precision conversion (bf16 staging)
  cvt_f32_bf16<<<dim3(CCH * CCH / 256), 256, 0, stream>>>(Wq, wqb, CCH * CCH);
  cvt_f32_bf16<<<dim3(CCH * CCH / 256), 256, 0, stream>>>(Wk, wkb, CCH * CCH);
  cvt_f32_bf16<<<dim3(CCH * CCH / 256), 256, 0, stream>>>(Wv, wvb, CCH * CCH);
  cvt_f32_bf16<<<dim3(CCH * CCH / 256), 256, 0, stream>>>(W1, w1b, CCH * CCH);
  cvt_f32_bf16<<<dim3(CCH * CCH / 256), 256, 0, stream>>>(W2, w2b, CCH * CCH);
  cvt_f32_bf16<<<dim3(PTOT * CCH / 256), 256, 0, stream>>>(x, xb, PTOT * CCH);

  // 2) spectral norm sigmas (1/sigma)
  spectral_sigma<<<1, 256, 0, stream>>>(W1, u1, rsig + 0);
  spectral_sigma<<<1, 256, 0, stream>>>(W2, u2, rsig + 1);

  // 3) fused Q/K/V projection (WMMA)
  gemm_cproj<0><<<dim3(PTOT / 16, 2), 256, 0, stream>>>(
      xb, wqb, wkb, wvb, bq, bk, bv, qb, kb, vbuf,
      rsig, beta, nullptr, nullptr);

  // 4) per-head energy + softmax (WMMA) -> attn (d_out tail) + bf16 copy
  attn_energy<<<dim3(BATCH * NHEAD), 128, 0, stream>>>(qb, kb, attn_out, attnb);

  // 5) attn @ v, gamma residual (WMMA) -> out1 (f32 + bf16)
  attn_apply<<<dim3(NPIX / 32, BATCH * NHEAD), 128, 0, stream>>>(
      attnb, vbuf, x, gamma, out1f, out1b);

  // 6) y1 = leaky_relu(W1/s1 @ out1 + b1) (WMMA)
  gemm_cproj<1><<<dim3(PTOT / 16, 2), 256, 0, stream>>>(
      out1b, w1b, w1b, w1b, b1, b1, b1, y1b, y1b, y1b,
      rsig + 0, beta, nullptr, nullptr);

  // 7) out = out1 + beta * (W2/s2 @ y1 + b2) (WMMA) -> d_out
  gemm_cproj<2><<<dim3(PTOT / 16, 2), 256, 0, stream>>>(
      y1b, w2b, w2b, w2b, b2, b2, b2, y1b, y1b, y1b,
      rsig + 1, beta, out1f, outf);
}
